// BSplineKANLayer_57483842290244
// MI455X (gfx1250) — matrix operand order repaired and checked by reading, not compiled, not run on hardware
//
#include <hip/hip_runtime.h>
#include <hip/hip_bf16.h>

// ---------------- problem constants (from reference) ----------------
#define N_TOK 8192
#define IN_F  1024
#define OUT_F 1024
#define NB    8            // B-spline basis functions per input feature
#define KPI   9            // expanded K per input feature: 1 silu + 8 basis
#define BM    128          // rows (tokens) per block tile
#define BN    128          // cols (out features) per block tile
#define KC    32           // input features per K chunk
#define KE    (KC * KPI)   // 288 expanded K per chunk (9 WMMA k-steps)
#define LDT   (KE + 8)     // 296 bf16 row stride in LDS (16B-aligned rows, de-conflicted)

typedef __attribute__((ext_vector_type(16))) __bf16       v16bf;
typedef __attribute__((ext_vector_type(8)))  float        v8f;
typedef __attribute__((ext_vector_type(4)))  unsigned int u32x4;

struct F32B { u32x4 lo, hi; };   // 32-byte carrier for a bf16 WMMA fragment

__device__ __forceinline__ unsigned int f2bf(float f) {
  unsigned int u = __float_as_uint(f);
  u += 0x7FFFu + ((u >> 16) & 1u);        // round-to-nearest-even
  return u >> 16;
}
__device__ __forceinline__ unsigned int pack2(float a, float b) {
  return f2bf(a) | (f2bf(b) << 16);
}

// Uniform-knot cubic B-spline: of the 8 bases, only 4 are non-zero for any x.
// Cell c = floor((x+1)/h) in 0..4, local u in [0,1); non-zero bases are
// j = c..c+3 with the cardinal cubic weights. Produces the 4 packed bf16-pair
// dwords for the 8 basis slots directly (slot 2j in low half of dword j).
__device__ __forceinline__ u32x4 bspl_pack(float x) {
  const float invh = 2.5f;                 // 1/h, h = 0.4
  float s  = (x + 1.0f) * invh;
  float cf = __builtin_fminf(__builtin_fmaxf(__builtin_floorf(s), 0.0f), 4.0f);
  float u  = s - cf;
  int   c  = (int)cf;

  float um  = 1.0f - u;
  float u2  = u * u;
  float u3  = u2 * u;
  const float i6 = 1.0f / 6.0f;
  float w0 = um * um * um * i6;                          // (1-u)^3/6
  float w1 = (3.0f * u3 - 6.0f * u2 + 4.0f) * i6;        // (3u^3-6u^2+4)/6
  float w2 = (-3.0f * u3 + 3.0f * u2 + 3.0f * u + 1.0f) * i6;
  float w3 = u3 * i6;                                    // u^3/6

  unsigned int h0 = f2bf(w0), h1 = f2bf(w1), h2 = f2bf(w2), h3 = f2bf(w3);
  int  e   = c >> 1;
  bool odd = (c & 1) != 0;
  // pair contents starting at pair index e:
  //   even cell: (w0,w1) (w2,w3)
  //   odd  cell: (0,w0)  (w1,w2) (w3,0)
  unsigned int a0 = odd ? (h0 << 16)        : (h0 | (h1 << 16));
  unsigned int a1 = odd ? (h1 | (h2 << 16)) : (h2 | (h3 << 16));
  unsigned int a2 = odd ? h3                : 0u;

  unsigned int r[4];
#pragma unroll
  for (int j = 0; j < 4; ++j) {
    int d = j - e;
    r[j] = (d == 0) ? a0 : (d == 1) ? a1 : (d == 2) ? a2 : 0u;
  }
  u32x4 pk;
  pk.x = r[0]; pk.y = r[1]; pk.z = r[2]; pk.w = r[3];
  return pk;
}

extern "C" __global__ __launch_bounds__(256, 2)
void kan_fused_bf16_wmma(const float* __restrict__ x,    // [N_TOK, IN_F]
                         const float* __restrict__ bw,   // [IN_F, OUT_F]
                         const float* __restrict__ sw,   // [OUT_F, IN_F, NB]
                         const float* __restrict__ ss,   // [OUT_F, IN_F]
                         float* __restrict__ out)        // [N_TOK, OUT_F]
{
  extern __shared__ unsigned short smem[];
  unsigned short* Al = smem;             // [BM][LDT]  expanded A tile (bf16 bits)
  unsigned short* Bl = smem + BM * LDT;  // [BN][LDT]  fused    B tile (bf16 bits)

  const int t    = threadIdx.x;
  const int lane = t & 31;
  const int wave = t >> 5;        // 8 waves
  const int wm   = wave >> 1;     // 0..3 : 32-row strip
  const int wn   = wave & 1;      // 0..1 : 64-col strip
  const int ln   = lane & 15;
  const int lh   = lane >> 4;

  const int brow = blockIdx.y * BM;
  const int bcol = blockIdx.x * BN;

  // fill-phase role: each thread owns (row fr, 16-feature half) of both tiles
  const int fr = t >> 1;
  const int i0 = (t & 1) * 16;

  v8f acc[2][4];
#pragma unroll
  for (int a = 0; a < 2; ++a)
#pragma unroll
    for (int b = 0; b < 4; ++b) { v8f z = {}; acc[a][b] = z; }

  for (int ch = 0; ch < IN_F / KC; ++ch) {
    const int ib = ch * KC;

    // ---------------- A tile: silu + closed-form spline expansion ----------------
    {
      const float* xp = x + (size_t)(brow + fr) * IN_F + ib + i0;
      float xv[16];
#pragma unroll
      for (int q = 0; q < 4; ++q) {
        float4 v = ((const float4*)xp)[q];
        xv[q*4+0] = v.x; xv[q*4+1] = v.y; xv[q*4+2] = v.z; xv[q*4+3] = v.w;
      }
      unsigned short* arow = Al + fr * LDT;
#pragma unroll
      for (int q = 0; q < 16; ++q) {
        const float xx = xv[q];
        // silu via v_rcp_f32
        float sl = xx * __builtin_amdgcn_rcpf(1.0f + __expf(-xx));
        arow[i0 + q] = (unsigned short)f2bf(sl);          // silu slot: k = i_local
        // basis slots: k = 32 + i_local*8 + b  (16B-aligned b128 store)
        *(u32x4*)(arow + KC + (i0 + q) * NB) = bspl_pack(xx);
      }
    }

    // ---------------- B tile: bw + sw*scaler fused into LDS ----------------
    {
      const int og = bcol + fr;
      float sc[16];
      const float4* scp = (const float4*)(ss + (size_t)og * IN_F + ib + i0);
#pragma unroll
      for (int q = 0; q < 4; ++q) {
        float4 v = scp[q];
        sc[q*4+0] = v.x; sc[q*4+1] = v.y; sc[q*4+2] = v.z; sc[q*4+3] = v.w;
      }
      unsigned short* brp = Bl + fr * LDT;
      const float4* swp = (const float4*)(sw + ((size_t)og * IN_F + ib + i0) * NB);
#pragma unroll
      for (int q = 0; q < 16; ++q) {
        float4 v0 = swp[q * 2 + 0];
        float4 v1 = swp[q * 2 + 1];
        const float s = sc[q];
        u32x4 pk;
        pk.x = pack2(v0.x * s, v0.y * s);
        pk.y = pack2(v0.z * s, v0.w * s);
        pk.z = pack2(v1.x * s, v1.y * s);
        pk.w = pack2(v1.z * s, v1.w * s);
        *(u32x4*)(brp + KC + (i0 + q) * NB) = pk;
      }
#pragma unroll
      for (int q = 0; q < 16; ++q)
        brp[i0 + q] = (unsigned short)f2bf(bw[(size_t)(ib + i0 + q) * OUT_F + og]);
    }

    __syncthreads();

    // ---------------- 9 WMMA k-steps over the 288-wide expanded chunk ----------------
#pragma unroll
    for (int ks = 0; ks < KE / 32; ++ks) {
      const int kb = ks * 32;
      v16bf af[2];
#pragma unroll
      for (int a = 0; a < 2; ++a) {
        // A 16x32 bf16 layout: lane half holds K in [8h,8h+8) U [16+8h,16+8h+8)
        const unsigned short* p = Al + (wm * 32 + a * 16 + ln) * LDT + kb + lh * 8;
        F32B f;
        f.lo = *(const u32x4*)p;
        f.hi = *(const u32x4*)(p + 16);
        af[a] = __builtin_bit_cast(v16bf, f);
      }
      v16bf bfr[4];
#pragma unroll
      for (int b = 0; b < 4; ++b) {
        // B 32x16 bf16 layout: lane half holds K in [8h,8h+8) U [16+8h,16+8h+8)
        const unsigned short* p = Bl + (wn * 64 + b * 16 + ln) * LDT + kb + lh * 8;
        F32B f;
        f.lo = *(const u32x4*)p;
        f.hi = *(const u32x4*)(p + 16);
        bfr[b] = __builtin_bit_cast(v16bf, f);
      }
#pragma unroll
      for (int a = 0; a < 2; ++a)
#pragma unroll
        for (int b = 0; b < 4; ++b)
          acc[a][b] = __builtin_amdgcn_wmma_f32_16x16x32_bf16(
              false, af[a], false, bfr[b], (short)0, acc[a][b], false, false);
    }

    __syncthreads();
  }

  // ---------------- epilogue: C/D layout -> global f32 ----------------
#pragma unroll
  for (int a = 0; a < 2; ++a) {
    const int rbase = brow + wm * 32 + a * 16 + lh * 8;   // VGPR j holds row rbase+j
#pragma unroll
    for (int b = 0; b < 4; ++b) {
      const int c = bcol + wn * 64 + b * 16 + ln;         // N = lane&15
#pragma unroll
      for (int j = 0; j < 8; ++j)
        out[(size_t)(rbase + j) * OUT_F + c] = acc[a][b][j];
    }
  }
}

extern "C" void kernel_launch(void* const* d_in, const int* in_sizes, int n_in,
                              void* d_out, int out_size, void* d_ws, size_t ws_size,
                              hipStream_t stream) {
  (void)in_sizes; (void)n_in; (void)out_size; (void)d_ws; (void)ws_size;
  const float* x  = (const float*)d_in[0];   // [8192,1024]
  const float* bw = (const float*)d_in[1];   // [1024,1024]
  const float* sw = (const float*)d_in[2];   // [1024,1024,8]
  const float* ss = (const float*)d_in[3];   // [1024,1024]
  float* out = (float*)d_out;

  dim3 grid(OUT_F / BN, N_TOK / BM);         // (8, 64)
  dim3 block(256);
  size_t lds_bytes = (size_t)2 * BM * LDT * sizeof(unsigned short);  // 151,552 B
  kan_fused_bf16_wmma<<<grid, block, lds_bytes, stream>>>(x, bw, sw, ss, out);
}
